// DynamicSparseMoE_51393578664658
// MI455X (gfx1250) — matrix operand order repaired and checked by reading, not compiled, not run on hardware
//
#include <hip/hip_runtime.h>
#include <math.h>

// ---------------- types ----------------
typedef __attribute__((ext_vector_type(16))) __bf16         v16bf;
typedef __attribute__((ext_vector_type(16))) unsigned short v16u;
typedef __attribute__((ext_vector_type(8)))  unsigned short v8u;
typedef __attribute__((ext_vector_type(8)))  float          v8f;
typedef __attribute__((ext_vector_type(4)))  float          v4f;

#define BB   8
#define TT   2048
#define CC   2048
#define EE   16
#define DEW  128          // per-expert input width
#define HHX  512          // expert hidden
#define NTOK (BB*TT)      // 16384 tokens
#define MT   32           // tokens per block in main kernel

union FragU { v16u u; v8u h[2]; };

// native f32 -> bf16 convert (lowers to v_cvt_*bf16_f32 on gfx1250)
__device__ __forceinline__ unsigned short f2bf(float f) {
    return __builtin_bit_cast(unsigned short, (__bf16)f);
}
__device__ __forceinline__ v16bf as_bf(v16u a) {
    return __builtin_bit_cast(v16bf, a);
}

// ---------------- 1) weight f32 -> bf16 conversion ----------------
__global__ void cvt_bf16_kernel(const float* __restrict__ src,
                                unsigned short* __restrict__ dst, int n) {
    int i = blockIdx.x * blockDim.x + threadIdx.x;
    if (i < n) dst[i] = f2bf(src[i]);
}

// ---------------- 2) gate: logits = x @ gate_w^T + gate_b ; mask = logits > 0 ----------------
// block = 128 threads (4 waves); each wave owns a 16-token x 16-expert tile, K = 2048.
__global__ __launch_bounds__(128) void gate_kernel(
    const float* __restrict__ x,                 // [NTOK][CC]
    const unsigned short* __restrict__ gwb,      // [EE][CC] bf16
    const float* __restrict__ gate_b,            // [EE]
    float* __restrict__ gate_out)                // [NTOK][EE]
{
    const int wave = threadIdx.x >> 5;
    const int lane = threadIdx.x & 31;
    const int hf   = lane >> 4;        // lane half
    const int lc   = lane & 15;
    const int t0   = (blockIdx.x * 4 + wave) * 16;

    v8f acc;
    const float bias = gate_b[lc];     // N (expert) = lc, same bias for all rows
    for (int r = 0; r < 8; ++r) acc[r] = bias;

    const float*          xrow = x   + (size_t)(t0 + lc) * CC;  // A row: token = lc
    const unsigned short* grow = gwb + (size_t)lc * CC;         // B col: expert = lc

    for (int ks = 0; ks < CC / 32; ++ks) {
        const int ka = ks * 32 + hf * 8;        // A: interleaved K groups
        const int kb = ks * 32 + hf * 8;        // B: interleaved K groups
        // A fragment: convert 16 fp32 -> bf16 in-register (native cvt)
        v4f a0 = *(const v4f*)(xrow + ka);
        v4f a1 = *(const v4f*)(xrow + ka + 4);
        v4f a2 = *(const v4f*)(xrow + ka + 16);
        v4f a3 = *(const v4f*)(xrow + ka + 20);
        FragU A;
        v8u lo, hi;
        for (int i = 0; i < 4; ++i) {
            lo[i] = f2bf(a0[i]); lo[i + 4] = f2bf(a1[i]);
            hi[i] = f2bf(a2[i]); hi[i + 4] = f2bf(a3[i]);
        }
        A.h[0] = lo; A.h[1] = hi;
        FragU Bf;
        Bf.h[0] = *(const v8u*)(grow + kb);
        Bf.h[1] = *(const v8u*)(grow + kb + 16);
        acc = __builtin_amdgcn_wmma_f32_16x16x32_bf16(
                  false, as_bf(A.u), false, as_bf(Bf.u), (short)0, acc, false, false);
    }
    // D tile: N = lc (expert), M = r + 8*hf
    for (int r = 0; r < 8; ++r) {
        int tok = t0 + r + 8 * hf;
        gate_out[tok * EE + lc] = (acc[r] > 0.0f) ? 1.0f : 0.0f;
    }
}

// ---------------- 3) per-expert MLP: fc -> gelu -> proj -> gate-scale ----------------
// grid = (NTOK/MT, EE), block = 256 threads (8 waves). LDS: X 8KB + H 32KB.
__global__ __launch_bounds__(256) void moe_kernel(
    const float* __restrict__ x,                 // [NTOK][CC]
    const unsigned short* __restrict__ fcw,      // [EE][HHX][DEW] bf16
    const float* __restrict__ fcb,               // [EE][HHX]
    const unsigned short* __restrict__ pjw,      // [EE][DEW][HHX] bf16
    const float* __restrict__ pjb,               // [EE][DEW]
    const float* __restrict__ gate,              // [NTOK][EE]
    float* __restrict__ out)                     // [NTOK][CC]
{
    __shared__ unsigned short Xs[MT * DEW];      // 8 KB
    __shared__ unsigned short Hs[MT * HHX];      // 32 KB

    const int e    = blockIdx.y;
    const int t0   = blockIdx.x * MT;
    const int tid  = threadIdx.x;
    const int wave = tid >> 5, lane = tid & 31;
    const int hf   = lane >> 4, lc = lane & 15;

    // ---- stage X tile (strided channel gather) as bf16: Xs[row][d] = x[t0+row][d*E + e]
    for (int i = 0; i < (MT * DEW) / 256; ++i) {
        int idx = i * 256 + tid;
        int row = idx >> 7, d = idx & (DEW - 1);
        Xs[idx] = f2bf(x[(size_t)(t0 + row) * CC + d * EE + e]);
    }
    __syncthreads();

    // ---- fc GEMM [MT x DEW] @ [DEW x HHX] + bias, exact GELU, stage H
    {
        const int wm = wave & 1;                 // 2 M tiles
        const int wn = wave >> 1;                // 4 wave columns x 8 ntiles
        FragU A[4];
        const unsigned short* xr = Xs + (wm * 16 + lc) * DEW;
        for (int ks = 0; ks < 4; ++ks) {
            int ka = ks * 32 + hf * 8;
            A[ks].h[0] = *(const v8u*)(xr + ka);
            A[ks].h[1] = *(const v8u*)(xr + ka + 16);
        }
        const unsigned short* fwe = fcw + (size_t)e * HHX * DEW;
        for (int nt = 0; nt < 8; ++nt) {
            int ntile = wn * 8 + nt;
            int hcol  = ntile * 16 + lc;         // hidden unit (N) for this lane
            const unsigned short* wr = fwe + (size_t)hcol * DEW;
            float bias = fcb[e * HHX + hcol];
            v8f acc;
            for (int r = 0; r < 8; ++r) acc[r] = bias;
            for (int ks = 0; ks < 4; ++ks) {
                int kb = ks * 32 + hf * 8;
                FragU Bf;
                Bf.h[0] = *(const v8u*)(wr + kb);
                Bf.h[1] = *(const v8u*)(wr + kb + 16);
                acc = __builtin_amdgcn_wmma_f32_16x16x32_bf16(
                          false, as_bf(A[ks].u), false, as_bf(Bf.u), (short)0, acc, false, false);
            }
            for (int r = 0; r < 8; ++r) {
                float v = acc[r];
                float g = 0.5f * v * (1.0f + erff(v * 0.70710678118654752f)); // exact gelu
                int row = wm * 16 + r + 8 * hf;
                Hs[row * HHX + hcol] = f2bf(g);
            }
        }
    }
    __syncthreads();

    // ---- proj GEMM [MT x HHX] @ [HHX x DEW] + bias, gate-scale, store
    {
        const int wm = wave & 1;                 // 2 M tiles
        const int wc = wave >> 1;                // 4 wave columns x 2 ntiles
        const unsigned short* hr  = Hs  + (wm * 16 + lc) * HHX;
        const unsigned short* pwe = pjw + (size_t)e * DEW * HHX;

        v8f acc[2];
        int dcol[2];
        for (int j = 0; j < 2; ++j) {
            int ntile = wc * 2 + j;
            dcol[j] = ntile * 16 + lc;           // output channel within expert
            float bias = pjb[e * DEW + dcol[j]];
            for (int r = 0; r < 8; ++r) acc[j][r] = bias;
        }
        for (int ks = 0; ks < HHX / 32; ++ks) {
            int ka = ks * 32 + hf * 8;
            int kb = ks * 32 + hf * 8;
            FragU A;
            A.h[0] = *(const v8u*)(hr + ka);
            A.h[1] = *(const v8u*)(hr + ka + 16);
            for (int j = 0; j < 2; ++j) {
                const unsigned short* wr = pwe + (size_t)dcol[j] * HHX;
                FragU Bf;
                Bf.h[0] = *(const v8u*)(wr + kb);
                Bf.h[1] = *(const v8u*)(wr + kb + 16);
                acc[j] = __builtin_amdgcn_wmma_f32_16x16x32_bf16(
                             false, as_bf(A.u), false, as_bf(Bf.u), (short)0, acc[j], false, false);
            }
        }
        for (int r = 0; r < 8; ++r) {
            int tok = t0 + wm * 16 + r + 8 * hf;
            float g = gate[tok * EE + e];
            float* orow = out + (size_t)tok * CC + e * DEW;   // c = e*DE + d layout
            for (int j = 0; j < 2; ++j)
                orow[dcol[j]] = g * acc[j][r];
        }
    }
}

// ---------------- launch ----------------
extern "C" void kernel_launch(void* const* d_in, const int* in_sizes, int n_in,
                              void* d_out, int out_size, void* d_ws, size_t ws_size,
                              hipStream_t stream) {
    (void)in_sizes; (void)n_in; (void)out_size; (void)ws_size;
    const float* x      = (const float*)d_in[0];
    const float* gate_w = (const float*)d_in[1];
    const float* gate_b = (const float*)d_in[2];
    const float* fc_w   = (const float*)d_in[3];
    const float* fc_b   = (const float*)d_in[4];
    const float* proj_w = (const float*)d_in[5];
    const float* proj_b = (const float*)d_in[6];
    float* out = (float*)d_out;

    char* ws = (char*)d_ws;
    // ws layout (256B aligned): gate_w bf16 | fc_w bf16 | proj_w bf16 | gate mask f32
    unsigned short* gwb   = (unsigned short*)(ws + 0);          //  65536 B
    unsigned short* fwb   = (unsigned short*)(ws + 65536);      // 2097152 B
    unsigned short* pwb   = (unsigned short*)(ws + 2162688);    // 2097152 B
    float*          gmask = (float*)         (ws + 4259840);    // 1048576 B

    // 1) weight conversion
    cvt_bf16_kernel<<<(EE * CC + 255) / 256, 256, 0, stream>>>(gate_w, gwb, EE * CC);
    cvt_bf16_kernel<<<(EE * HHX * DEW + 255) / 256, 256, 0, stream>>>(fc_w, fwb, EE * HHX * DEW);
    cvt_bf16_kernel<<<(EE * DEW * HHX + 255) / 256, 256, 0, stream>>>(proj_w, pwb, EE * DEW * HHX);

    // 2) gate mask
    gate_kernel<<<NTOK / 64, 128, 0, stream>>>(x, gwb, gate_b, gmask);

    // 3) expert MLPs + combine
    dim3 grid(NTOK / MT, EE);
    moe_kernel<<<grid, 256, 0, stream>>>(x, fwb, fc_b, pwb, proj_b, gmask, out);
}
